// RNNennigma_1236950582097
// MI455X (gfx1250) — matrix operand order repaired and checked by reading, not compiled, not run on hardware
//
#include <hip/hip_runtime.h>
#include <hip/hip_bf16.h>

#define DIM 64
#define TWO_D 128
#define NSYM 8
#define NL1 65536
#define NL2 32768
#define NCLAUSE 2048
#define TLIT 16

typedef _Float16 v16h_t __attribute__((ext_vector_type(16)));
typedef float    v8f_t  __attribute__((ext_vector_type(8)));
typedef unsigned int v4u_t __attribute__((ext_vector_type(4)));
typedef int v8i_t __attribute__((ext_vector_type(8)));
typedef int v4i_t __attribute__((ext_vector_type(4)));

union AF { v16h_t v; _Float16 h[16]; uint4 q[2]; };
union CF { v8f_t  v; float    f[8];  };

#if __has_builtin(__builtin_amdgcn_tensor_load_to_lds)
#define USE_TDM 1
#else
#define USE_TDM 0
#endif

// ---- Tensor Data Mover: 1-D contiguous copy of nhalf f16 elements to LDS ----
// D# per CDNA5 ISA ch.8: group0 {count=1, lds_addr, global_addr, type=2},
// group1 {data_size=2B, tensor_dim0=nhalf, tensor_dim1=1, tile_dim0=nhalf}.
__device__ __forceinline__ void tdm_copy_half(void* lds_dst, const void* gsrc,
                                              unsigned nhalf) {
#if USE_TDM
  unsigned long long ga = (unsigned long long)(uintptr_t)gsrc;
  v4u_t g0;
  g0.x = 1u;                                   // count=1 (valid user descriptor)
  g0.y = (unsigned)(uintptr_t)lds_dst;         // lds_addr (generic low32 = LDS off)
  g0.z = (unsigned)ga;                         // global_addr[31:0]
  g0.w = (unsigned)((ga >> 32) & 0x01ffffffu)  // global_addr[56:32]
         | (2u << 30);                         // type=2 ("image")
  v8i_t g1;
  g1[0] = (int)(1u << 16);                     // data_size=1 -> 2 bytes
  g1[1] = (int)((nhalf & 0xffffu) << 16);      // tensor_dim0[15:0]
  g1[2] = (int)(((nhalf >> 16) & 0xffffu) | (1u << 16));  // td0[31:16], tensor_dim1=1
  g1[3] = (int)((nhalf & 0xffffu) << 16);      // tile_dim0 = nhalf (<=65535)
  g1[4] = 0;                                   // tile_dim1=0, tile_dim2=0
  g1[5] = (int)nhalf;                          // tensor_dim0_stride
  g1[6] = 0;
  g1[7] = 0;
  v4i_t z;
  z[0] = 0; z[1] = 0; z[2] = 0; z[3] = 0;
#if defined(__clang_major__) && __clang_major__ >= 23
  v8i_t z8;
  z8[0] = 0; z8[1] = 0; z8[2] = 0; z8[3] = 0;
  z8[4] = 0; z8[5] = 0; z8[6] = 0; z8[7] = 0;
  __builtin_amdgcn_tensor_load_to_lds(g0, g1, z, z, z8, 0);
#else
  __builtin_amdgcn_tensor_load_to_lds(g0, g1, z, z, 0);
#endif
#else
  (void)lds_dst; (void)gsrc; (void)nhalf;
#endif
}

// A/B fragment K patterns (wave32, 16-bit operands):
//   A: lanes 0-15 hold K = {kb+0..7, kb+16..23}; lanes 16-31 K = {kb+8..15, kb+24..31}
//   B: same K pattern as A (col = lane&15)

// 16 contiguous halves, 16B-aligned base -> two ds_load_b128
__device__ __forceinline__ v16h_t frag_contig(const _Float16* p) {
  AF a;
  const uint4* q = (const uint4*)p;
  a.q[0] = q[0];
  a.q[1] = q[1];
  return a.v;
}
// two 8-half chunks at p and p+16 halves
__device__ __forceinline__ v16h_t frag_split(const _Float16* p) {
  AF a;
  a.q[0] = *(const uint4*)p;
  a.q[1] = *(const uint4*)(p + 16);
  return a.v;
}
// 8 contiguous f32 -> 8 halves (two global_load_b128 + packed cvt)
__device__ __forceinline__ void cvt8(const float* p, _Float16* d) {
  float4 x = ((const float4*)p)[0];
  float4 y = ((const float4*)p)[1];
  d[0] = (_Float16)x.x; d[1] = (_Float16)x.y;
  d[2] = (_Float16)x.z; d[3] = (_Float16)x.w;
  d[4] = (_Float16)y.x; d[5] = (_Float16)y.y;
  d[6] = (_Float16)y.z; d[7] = (_Float16)y.w;
}

__device__ __forceinline__ float sigmoidf_(float x) {
  return 1.0f / (1.0f + expf(-x));
}

// ---------------------------------------------------------------- bucketing
__global__ void zero_counts_kernel(int* cnt) {
  if (threadIdx.x < 16) cnt[threadIdx.x] = 0;
}

__global__ void bucketize_kernel(const int* __restrict__ sym, int* cnt,
                                 int* __restrict__ bkt, int n, int stride) {
  int i = blockIdx.x * blockDim.x + threadIdx.x;
  if (i < n) {
    int s = sym[i];
    int p = atomicAdd(&cnt[s], 1);
    bkt[s * stride + p] = i;
  }
}

// --------------------------------------------- weight pre-transpose/convert
// W1 [S][k=128][c=128] f32 -> W1t [S][c][k] f16 ; W2 [S][k=128][c=64] -> [S][c][128]
// Wih/Whh [2][gcol=256][d=64] f32 -> f16 (layout kept; contiguous in K already)
__global__ void prep_weights_kernel(const float* __restrict__ W1,
                                    const float* __restrict__ W2,
                                    const float* __restrict__ Wih,
                                    const float* __restrict__ Whh,
                                    _Float16* __restrict__ W1t,
                                    _Float16* __restrict__ W2t,
                                    _Float16* __restrict__ WihH,
                                    _Float16* __restrict__ WhhH) {
  const int n1 = NSYM * TWO_D * TWO_D;  // 131072
  const int n2 = NSYM * DIM * TWO_D;    // 65536
  const int n3 = 2 * 4 * DIM * DIM;     // 32768
  int i = blockIdx.x * blockDim.x + threadIdx.x;
  if (i < n1) {
    int s = i >> 14, r = i & 16383, k = r >> 7, c = r & 127;
    W1t[s * 16384 + c * TWO_D + k] = (_Float16)W1[i];
  } else if (i < n1 + n2) {
    int j = i - n1;
    int s = j >> 13, r = j & 8191, k = r >> 6, c = r & 63;
    W2t[s * 8192 + c * TWO_D + k] = (_Float16)W2[j];
  } else if (i < n1 + n2 + n3) {
    int j = i - n1 - n2;
    WihH[j] = (_Float16)Wih[j];
  } else if (i < n1 + n2 + 2 * n3) {
    int j = i - n1 - n2 - n3;
    WhhH[j] = (_Float16)Whh[j];
  }
}

// --------------------------------------------------- per-symbol combiner MLP
// One wave computes 16 nodes of one symbol bucket:
//   Y1[16,128] = relu(X[16,128] @ W1[s] + b1[s])
//   Y2[16, 64] = relu(Y1       @ W2[s] + b2[s])
// Pre-transposed f16 weights DMA'd to LDS via TDM (B fragments contiguous).
__global__ __launch_bounds__(128) void combine_kernel(
    const float* __restrict__ s_emb, const int* __restrict__ leaf_ids,
    const float* __restrict__ xsrc, int mode,
    const int* __restrict__ left, const int* __restrict__ right,
    const _Float16* __restrict__ W1t, const float* __restrict__ b1,
    const _Float16* __restrict__ W2t, const float* __restrict__ b2,
    const int* __restrict__ bkt, const int* __restrict__ cnt, int nmax,
    float* __restrict__ out) {
  __shared__ __align__(16) _Float16 sW1t[TWO_D * TWO_D];  // [col][k], 32 KB
  __shared__ __align__(16) _Float16 sW2t[DIM * TWO_D];    // [col][k], 16 KB
  __shared__ __align__(16) _Float16 sY1[4][16 * TWO_D];   // [m][k],   16 KB

  const int sym = blockIdx.y;
  const int n = cnt[sym];
  if ((int)(blockIdx.x * 64) >= n) return;

  const int tid = threadIdx.x;
#if USE_TDM
  if (tid < 32) {
    tdm_copy_half(sW1t, W1t + sym * TWO_D * TWO_D, TWO_D * TWO_D);
    tdm_copy_half(sW2t, W2t + sym * DIM * TWO_D, DIM * TWO_D);
    __builtin_amdgcn_s_wait_tensorcnt(0);
  }
#else
  {
    const uint4* s1 = (const uint4*)(W1t + sym * TWO_D * TWO_D);
    uint4* d1 = (uint4*)sW1t;
    for (int i = tid; i < TWO_D * TWO_D / 8; i += 128) d1[i] = s1[i];
    const uint4* s2 = (const uint4*)(W2t + sym * DIM * TWO_D);
    uint4* d2 = (uint4*)sW2t;
    for (int i = tid; i < DIM * TWO_D / 8; i += 128) d2[i] = s2[i];
  }
#endif
  __syncthreads();

  const int wave = tid >> 5;
  const int lane = tid & 31;
  const int hi = lane >> 4;
  const int lcol = lane & 15;
  const int row0 = (blockIdx.x * 4 + wave) * 16;
  const bool active = row0 < n;

  if (active) {
    int ridx = row0 + lcol;
    if (ridx > n - 1) ridx = n - 1;
    int node = bkt[sym * nmax + ridx];
    const float* xl;
    const float* xr;
    if (mode == 0) {
      xl = s_emb + (long)leaf_ids[left[node]] * DIM;
      xr = s_emb + (long)leaf_ids[right[node]] * DIM;
    } else {
      xl = xsrc + (long)left[node] * DIM;
      xr = xsrc + (long)right[node] * DIM;
    }

    CF acc[8];
#pragma unroll
    for (int nt = 0; nt < 8; ++nt)
#pragma unroll
      for (int j = 0; j < 8; ++j) acc[nt].f[j] = 0.0f;

#pragma unroll
    for (int kc = 0; kc < 4; ++kc) {
      AF a;
      int b0 = kc * 32 + (hi ? 8 : 0);
      int b1_ = b0 + 16;
      const float* p0 = (b0 < DIM) ? (xl + b0) : (xr + b0 - DIM);
      const float* p1 = (b1_ < DIM) ? (xl + b1_) : (xr + b1_ - DIM);
      cvt8(p0, a.h);
      cvt8(p1, a.h + 8);
      const int kb = kc * 32 + (hi ? 8 : 0);
#pragma unroll
      for (int nt = 0; nt < 8; ++nt) {
        v16h_t b = frag_split(&sW1t[(nt * 16 + lcol) * TWO_D + kb]);
        acc[nt].v = __builtin_amdgcn_wmma_f32_16x16x32_f16(
            false, a.v, false, b, (short)0, acc[nt].v, false, false);
      }
    }
    _Float16* y = &sY1[wave][0];
#pragma unroll
    for (int nt = 0; nt < 8; ++nt) {
      int colb = nt * 16 + lcol;
      float bias = b1[sym * TWO_D + colb];
#pragma unroll
      for (int j = 0; j < 8; ++j) {
        float v = acc[nt].f[j] + bias;
        v = v > 0.0f ? v : 0.0f;
        y[(j + hi * 8) * TWO_D + colb] = (_Float16)v;
      }
    }
  }
  __syncthreads();
  if (active) {
    const _Float16* y = &sY1[wave][0];
    CF acc2[4];
#pragma unroll
    for (int nt = 0; nt < 4; ++nt)
#pragma unroll
      for (int j = 0; j < 8; ++j) acc2[nt].f[j] = 0.0f;

#pragma unroll
    for (int kc = 0; kc < 4; ++kc) {
      v16h_t a = frag_split(&y[lcol * TWO_D + kc * 32 + (hi ? 8 : 0)]);
      const int kb = kc * 32 + (hi ? 8 : 0);
#pragma unroll
      for (int nt = 0; nt < 4; ++nt) {
        v16h_t b = frag_split(&sW2t[(nt * 16 + lcol) * TWO_D + kb]);
        acc2[nt].v = __builtin_amdgcn_wmma_f32_16x16x32_f16(
            false, a, false, b, (short)0, acc2[nt].v, false, false);
      }
    }
#pragma unroll
    for (int nt = 0; nt < 4; ++nt) {
      int colb = nt * 16 + lcol;
      float bias = b2[sym * DIM + colb];
#pragma unroll
      for (int j = 0; j < 8; ++j) {
        int ridx = row0 + j + hi * 8;
        if (ridx < n) {
          int node = bkt[sym * nmax + ridx];
          float v = acc2[nt].f[j] + bias;
          out[(long)node * DIM + colb] = v > 0.0f ? v : 0.0f;
        }
      }
    }
  }
}

// ------------------------------------------------------------------- LSTM
// One wave owns 16 clauses. Per step: gates[16,256] = x@Wih^T + h@Whh^T + b,
// as 4(dt) x 4(gate) 16x16 WMMA tiles with K=64 (2 chunks) per matrix.
__global__ __launch_bounds__(128) void lstm_kernel(
    const float* __restrict__ xsrc, const int* __restrict__ clause_lits,
    int layer, const _Float16* __restrict__ WihH,
    const _Float16* __restrict__ WhhH, const float* __restrict__ bih,
    const float* __restrict__ bhh, float* __restrict__ hout) {
  __shared__ __align__(16) _Float16 sWih[4 * DIM * DIM];  // 32 KB [gcol][d]
  __shared__ __align__(16) _Float16 sWhh[4 * DIM * DIM];  // 32 KB
  __shared__ __align__(16) _Float16 sH[4][2][16 * DIM];   // 16 KB [m][d]

  const int tid = threadIdx.x;
#if USE_TDM
  if (tid < 32) {
    tdm_copy_half(sWih, WihH, 4 * DIM * DIM);
    tdm_copy_half(sWhh, WhhH, 4 * DIM * DIM);
    __builtin_amdgcn_s_wait_tensorcnt(0);
  }
#else
  {
    const uint4* s1 = (const uint4*)WihH;
    const uint4* s2 = (const uint4*)WhhH;
    uint4* d1 = (uint4*)sWih;
    uint4* d2 = (uint4*)sWhh;
    for (int i = tid; i < 4 * DIM * DIM / 8; i += 128) {
      d1[i] = s1[i];
      d2[i] = s2[i];
    }
  }
#endif
  const int wave = tid >> 5;
  const int lane = tid & 31;
  const int hi = lane >> 4;
  const int lcol = lane & 15;
  const int clause0 = (blockIdx.x * 4 + wave) * 16;

  for (int i = lane; i < 16 * DIM; i += 32) sH[wave][0][i] = (_Float16)0.0f;

  float bias[4][4];
#pragma unroll
  for (int dt = 0; dt < 4; ++dt)
#pragma unroll
    for (int gi = 0; gi < 4; ++gi) {
      int col = gi * DIM + dt * 16 + lcol;
      bias[dt][gi] = bih[col] + bhh[col];
    }

  CF cst[4];
#pragma unroll
  for (int dt = 0; dt < 4; ++dt)
#pragma unroll
    for (int j = 0; j < 8; ++j) cst[dt].f[j] = 0.0f;

  __syncthreads();

  const int m = lcol;  // this lane's A-fragment row
  for (int t = 0; t < TLIT; ++t) {
    const int cur = t & 1, nxt = cur ^ 1;
    const float* xrow;
    if (layer == 0) {
      int lit = clause_lits[(clause0 + m) * TLIT + t];
      xrow = xsrc + (long)lit * DIM;
    } else {
      xrow = xsrc + ((long)(clause0 + m) * TLIT + t) * DIM;
    }
    AF ax[2];
    v16h_t ah[2];
#pragma unroll
    for (int kc = 0; kc < 2; ++kc) {
      int b0 = kc * 32 + (hi ? 8 : 0);
      cvt8(xrow + b0, ax[kc].h);
      cvt8(xrow + b0 + 16, ax[kc].h + 8);
      ah[kc] = frag_split(&sH[wave][cur][m * DIM + b0]);
    }
#pragma unroll
    for (int dt = 0; dt < 4; ++dt) {
      CF g[4];
#pragma unroll
      for (int gi = 0; gi < 4; ++gi) {
#pragma unroll
        for (int j = 0; j < 8; ++j) g[gi].f[j] = 0.0f;
        int col = gi * DIM + dt * 16 + lcol;
#pragma unroll
        for (int kc = 0; kc < 2; ++kc) {
          int kb = kc * 32 + (hi ? 8 : 0);
          v16h_t b0 = frag_split(&sWih[col * DIM + kb]);
          g[gi].v = __builtin_amdgcn_wmma_f32_16x16x32_f16(
              false, ax[kc].v, false, b0, (short)0, g[gi].v, false, false);
          v16h_t b1 = frag_split(&sWhh[col * DIM + kb]);
          g[gi].v = __builtin_amdgcn_wmma_f32_16x16x32_f16(
              false, ah[kc], false, b1, (short)0, g[gi].v, false, false);
        }
      }
#pragma unroll
      for (int j = 0; j < 8; ++j) {
        float iv = sigmoidf_(g[0].f[j] + bias[dt][0]);
        float fv = sigmoidf_(g[1].f[j] + bias[dt][1]);
        float gv = tanhf(g[2].f[j] + bias[dt][2]);
        float ov = sigmoidf_(g[3].f[j] + bias[dt][3]);
        float cc = fv * cst[dt].f[j] + iv * gv;
        cst[dt].f[j] = cc;
        float hv = ov * tanhf(cc);
        int rowm = j + hi * 8;
        int col = dt * 16 + lcol;
        sH[wave][nxt][rowm * DIM + col] = (_Float16)hv;
        if (layer == 0) {
          hout[((long)(clause0 + rowm) * TLIT + t) * DIM + col] = hv;
        } else if (t == TLIT - 1) {
          hout[(long)(clause0 + rowm) * DIM + col] = hv;
        }
      }
    }
    __syncthreads();
  }
}

// ------------------------------------------------------------------ scorer
__global__ void scorer_kernel(const float* __restrict__ hlast,
                              const float* __restrict__ Wf1,
                              const float* __restrict__ bf1,
                              const float* __restrict__ Wf2,
                              const float* __restrict__ bf2,
                              float* __restrict__ out) {
  int c = blockIdx.x * blockDim.x + threadIdx.x;
  if (c >= NCLAUSE) return;
  const float* h = hlast + (long)c * DIM;
  float hb[DIM];
  for (int d = 0; d < DIM; ++d) hb[d] = h[d];
  float o0 = bf2[0], o1 = bf2[1];
  for (int j = 0; j < 32; ++j) {
    float a = bf1[j];
    for (int d = 0; d < DIM; ++d) a += hb[d] * Wf1[d * 32 + j];
    a = a > 0.0f ? a : 0.0f;
    o0 += a * Wf2[j * 2 + 0];
    o1 += a * Wf2[j * 2 + 1];
  }
  out[c * 2 + 0] = o0;
  out[c * 2 + 1] = o1;
}

// ---------------------------------------------------------------- launcher
extern "C" void kernel_launch(void* const* d_in, const int* in_sizes, int n_in,
                              void* d_out, int out_size, void* d_ws,
                              size_t ws_size, hipStream_t stream) {
  const float* s_emb = (const float*)d_in[0];
  const float* W1 = (const float*)d_in[1];
  const float* b1 = (const float*)d_in[2];
  const float* W2 = (const float*)d_in[3];
  const float* b2 = (const float*)d_in[4];
  const float* Wih = (const float*)d_in[5];
  const float* Whh = (const float*)d_in[6];
  const float* bih = (const float*)d_in[7];
  const float* bhh = (const float*)d_in[8];
  // d_in[9]=h0, d_in[10]=c0 are zeros (used as literal 0 in kernels)
  const float* Wf1 = (const float*)d_in[11];
  const float* bf1 = (const float*)d_in[12];
  const float* Wf2 = (const float*)d_in[13];
  const float* bf2 = (const float*)d_in[14];
  const int* leaf_ids = (const int*)d_in[15];
  const int* l1_sym = (const int*)d_in[16];
  const int* l1_left = (const int*)d_in[17];
  const int* l1_right = (const int*)d_in[18];
  const int* l2_sym = (const int*)d_in[19];
  const int* l2_left = (const int*)d_in[20];
  const int* l2_right = (const int*)d_in[21];
  const int* clause_lits = (const int*)d_in[22];
  float* out = (float*)d_out;

  char* ws = (char*)d_ws;
  size_t off = 0;
  float* n1 = (float*)(ws + off);    off += (size_t)NL1 * DIM * 4;
  float* lits = (float*)(ws + off);  off += (size_t)NL2 * DIM * 4;
  float* hseq = (float*)(ws + off);  off += (size_t)NCLAUSE * TLIT * DIM * 4;
  float* hlast = (float*)(ws + off); off += (size_t)NCLAUSE * DIM * 4;
  int* cnt = (int*)(ws + off);       off += 64;
  int* bkt1 = (int*)(ws + off);      off += (size_t)NSYM * NL1 * 4;
  int* bkt2 = (int*)(ws + off);      off += (size_t)NSYM * NL2 * 4;
  _Float16* W1t = (_Float16*)(ws + off);  off += (size_t)NSYM * TWO_D * TWO_D * 2;
  _Float16* W2t = (_Float16*)(ws + off);  off += (size_t)NSYM * DIM * TWO_D * 2;
  _Float16* WihH = (_Float16*)(ws + off); off += (size_t)2 * 4 * DIM * DIM * 2;
  _Float16* WhhH = (_Float16*)(ws + off); off += (size_t)2 * 4 * DIM * DIM * 2;
  (void)off; (void)ws_size; (void)n_in; (void)in_sizes; (void)out_size;

  zero_counts_kernel<<<1, 32, 0, stream>>>(cnt);
  prep_weights_kernel<<<(NSYM * TWO_D * TWO_D + NSYM * DIM * TWO_D +
                         2 * 2 * 4 * DIM * DIM + 255) / 256,
                        256, 0, stream>>>(W1, W2, Wih, Whh, W1t, W2t, WihH,
                                          WhhH);
  bucketize_kernel<<<(NL1 + 255) / 256, 256, 0, stream>>>(l1_sym, cnt, bkt1,
                                                          NL1, NL1);
  bucketize_kernel<<<(NL2 + 255) / 256, 256, 0, stream>>>(l2_sym, cnt + 8,
                                                          bkt2, NL2, NL2);

  combine_kernel<<<dim3(NL1 / 64, NSYM), 128, 0, stream>>>(
      s_emb, leaf_ids, (const float*)nullptr, 0, l1_left, l1_right, W1t, b1,
      W2t, b2, bkt1, cnt, NL1, n1);
  combine_kernel<<<dim3(NL2 / 64, NSYM), 128, 0, stream>>>(
      s_emb, leaf_ids, n1, 1, l2_left, l2_right, W1t, b1, W2t, b2, bkt2,
      cnt + 8, NL2, lits);

  lstm_kernel<<<NCLAUSE / 64, 128, 0, stream>>>(lits, clause_lits, 0, WihH,
                                                WhhH, bih, bhh, hseq);
  lstm_kernel<<<NCLAUSE / 64, 128, 0, stream>>>(
      hseq, clause_lits, 1, WihH + 4 * DIM * DIM, WhhH + 4 * DIM * DIM,
      bih + 4 * DIM, bhh + 4 * DIM, hlast);

  scorer_kernel<<<(NCLAUSE + 127) / 128, 128, 0, stream>>>(hlast, Wf1, bf1,
                                                           Wf2, bf2, out);
}
